// BeeSender_49057116454978
// MI455X (gfx1250) — compile-verified, matrix operand order checked by reading, not run on hardware
//
#include <hip/hip_runtime.h>
#include <hip/hip_bf16.h>

typedef __attribute__((ext_vector_type(16))) __bf16 v16bf;
typedef __attribute__((ext_vector_type(8)))  __bf16 v8bf;
typedef __attribute__((ext_vector_type(4)))  __bf16 v4bf;
typedef __attribute__((ext_vector_type(8)))  float  v8f;

#define NN   100000
#define EE   1600000
#define RR   4
#define FF   128
#define EMBD 128
#define HIDD 256
#define BB   1024

static __device__ __forceinline__ void atomic_add_f32(float* p, float v) {
    __hip_atomic_fetch_add(p, v, __ATOMIC_RELAXED, __HIP_MEMORY_SCOPE_AGENT);
}

// ---------------------------------------------------------------------------
// fp32 -> bf16 convert (whole array)
// ---------------------------------------------------------------------------
__global__ __launch_bounds__(256)
void cvt_bf16(const float* __restrict__ src, __bf16* __restrict__ dst, long n) {
    long i = (long)blockIdx.x * blockDim.x + threadIdx.x;
    if (i < n) dst[i] = (__bf16)src[i];
}

// ---------------------------------------------------------------------------
// relu + fp32 -> bf16 convert (layer-1 output -> layer-2 input)
// ---------------------------------------------------------------------------
__global__ __launch_bounds__(256)
void relu_cvt_bf16(const float* __restrict__ src, __bf16* __restrict__ dst, long n) {
    long i = (long)blockIdx.x * blockDim.x + threadIdx.x;
    if (i < n) dst[i] = (__bf16)fmaxf(src[i], 0.0f);
}

// ---------------------------------------------------------------------------
// Zero-fill fp32
// ---------------------------------------------------------------------------
__global__ __launch_bounds__(256)
void zero_f32(float* __restrict__ p, long n) {
    long i = (long)blockIdx.x * blockDim.x + threadIdx.x;
    if (i < n) p[i] = 0.0f;
}

// ---------------------------------------------------------------------------
// Degree count per (dst, relation) key
// ---------------------------------------------------------------------------
__global__ __launch_bounds__(256)
void count_deg(const int* __restrict__ edge_index,
               const int* __restrict__ edge_type,
               float* __restrict__ deg) {
    long e = (long)blockIdx.x * blockDim.x + threadIdx.x;
    if (e >= EE) return;
    int dst = edge_index[EE + e];
    int r   = edge_type[e];
    atomic_add_f32(&deg[(size_t)dst * RR + r], 1.0f);
}

// ---------------------------------------------------------------------------
// Pack a fp32 [K, NC] weight matrix into bf16 WMMA B-fragments:
//   dst[((nt*ksteps + ks)*32 + lane)*16 + e] = (bf16) src[ kk*NC + col ]
// with kk = ks*32 + 8*(lane>>4) + (e&7) + (e>=8 ? 16 : 0), col = nt*16+(lane&15)
// so each lane's 16-element fragment is 32B contiguous at run time.
// ---------------------------------------------------------------------------
__global__ __launch_bounds__(256)
void pack_wmma_b(const float* __restrict__ src, __bf16* __restrict__ dst,
                 int ksteps, int NC, long total) {
    long i = (long)blockIdx.x * blockDim.x + threadIdx.x;
    if (i >= total) return;
    const int e    = (int)(i & 15);
    const int lane = (int)((i >> 4) & 31);
    const long fr  = i >> 9;                 // fragment index = nt*ksteps + ks
    const int ks   = (int)(fr % ksteps);
    const int nt   = (int)(fr / ksteps);
    const int kk   = ks * 32 + ((lane >> 4) * 8) + (e & 7) + ((e >= 8) ? 16 : 0);
    const int col  = nt * 16 + (lane & 15);
    dst[i] = (__bf16)src[(size_t)kk * NC + col];
}

// ---------------------------------------------------------------------------
// Per-relation transform:  H[n, rel, :] = bf16( XB[n,:] @ W[rel] )
// Fused root pass (blockIdx.y == RR):  OUT[n,:] = XB[n,:] @ W[R] + b  (fp32)
//
// A tile (16 x 128 bf16, 4KB) staged to LDS via the CDNA5 async path
// (global_load_async_to_lds_b128 / s_wait_asynccnt), fragments read with
// ds_load_b128 (272B row pitch -> disjoint bank groups).
// B fragments come pre-swizzled (PB) -> one 32B contiguous load per k-step.
// ---------------------------------------------------------------------------
__global__ __launch_bounds__(256)
void rgcn_transform(const __bf16* __restrict__ XB,   // [N, 128] bf16
                    const __bf16* __restrict__ PB,   // [R+1][8nt][4ks][32][16] bf16
                    const float* __restrict__ bias,  // [128]
                    __bf16* __restrict__ H,          // [N, R, 128] bf16
                    float* __restrict__ OUT) {       // [N, 128] fp32
    __shared__ __align__(16) __bf16 smA[16 * 136];

    const int lane = threadIdx.x & 31;
    const int wave = threadIdx.x >> 5;      // n-tile 0..7
    const int m0   = blockIdx.x * 16;
    const int rel  = blockIdx.y;            // 0..3 relations, 4 = root
    const int lh   = lane & 15;
    const int kb   = (lane >> 4) * 8;
    const int ncol = wave * 16 + lh;

    // ---- async stage of the A tile: 256 lanes x 16B = 4KB ----
    {
        const int row   = threadIdx.x >> 4;   // 0..15
        const int chunk = threadIdx.x & 15;   // 16B chunks along the row
        unsigned ldsoff = (unsigned)(unsigned long long)&smA[row * 136 + chunk * 8];
        const __bf16* gp = XB + ((size_t)(m0 + row) * 128 + chunk * 8);
        asm volatile("global_load_async_to_lds_b128 %0, %1, off"
                     :: "v"(ldsoff), "v"(gp) : "memory");
        asm volatile("s_wait_asynccnt 0x0" ::: "memory");
    }
    __syncthreads();

    // lane's pre-swizzled B fragments for this (rel, n-tile)
    const __bf16* pb = PB + (size_t)rel * (128 * 128)
                          + (size_t)wave * (4 * 512)
                          + (size_t)lane * 16;

    v8f acc = {};
    #pragma unroll
    for (int ks = 0; ks < 4; ++ks) {
        const int k0 = ks * 32;
        const v8bf r1 = *(const v8bf*)&smA[lh * 136 + (k0 + kb)];
        const v8bf r2 = *(const v8bf*)&smA[lh * 136 + (k0 + kb + 16)];
        const v16bf a = __builtin_shufflevector(r1, r2,
                            0, 1, 2, 3, 4, 5, 6, 7, 8, 9, 10, 11, 12, 13, 14, 15);
        const v16bf b = *(const v16bf*)(pb + (size_t)ks * 512);
        acc = __builtin_amdgcn_wmma_f32_16x16x32_bf16(false, a, false, b,
                                                      (short)0, acc, false, false);
    }

    if (rel < RR) {
        #pragma unroll
        for (int r = 0; r < 8; ++r) {
            int m = m0 + r + kb;
            H[((size_t)m * RR + rel) * 128 + ncol] = (__bf16)acc[r];
        }
    } else {
        float bv = bias[ncol];
        #pragma unroll
        for (int r = 0; r < 8; ++r) {
            int m = m0 + r + kb;
            OUT[(size_t)m * 128 + ncol] = acc[r] + bv;
        }
    }
}

// ---------------------------------------------------------------------------
// Edge phase: one wave per edge. Edge indices are wave-uniform -> force them
// into SGPRs (readfirstlane) so gathers/atomics use saddr+voffset addressing.
// msg = H_bf16[src, etype, :] * (1/max(deg(dst,etype),1));  OUT[dst,:] += msg
// ---------------------------------------------------------------------------
__global__ __launch_bounds__(256)
void edge_messages(const int* __restrict__ edge_index,
                   const int* __restrict__ edge_type,
                   const __bf16* __restrict__ H,     // [N, R, 128]
                   const float* __restrict__ deg,    // [N*R]
                   float* __restrict__ OUT) {        // [N, 128]
    const int lane = threadIdx.x & 31;
    const int wave = threadIdx.x >> 5;
    const int e = __builtin_amdgcn_readfirstlane(blockIdx.x * 8 + wave);
    if (e >= EE) return;
    const int src = __builtin_amdgcn_readfirstlane(edge_index[e]);
    const int dst = __builtin_amdgcn_readfirstlane(edge_index[EE + e]);
    const int r   = __builtin_amdgcn_readfirstlane(edge_type[e]);
    const float d    = deg[(size_t)dst * RR + r];
    const float norm = 1.0f / fmaxf(d, 1.0f);

    const v4bf m = *(const v4bf*)(H + ((size_t)src * RR + r) * 128 + lane * 4);
    float* o = OUT + (size_t)dst * 128 + lane * 4;
    atomic_add_f32(o + 0, (float)m[0] * norm);
    atomic_add_f32(o + 1, (float)m[1] * norm);
    atomic_add_f32(o + 2, (float)m[2] * norm);
    atomic_add_f32(o + 3, (float)m[3] * norm);
}

// ---------------------------------------------------------------------------
// FC head: out[b,:] = relu( [h[nest[b]], h[food[b]]] @ W_fc + b_fc )
// Gathered A rows (fp32->bf16 on the fly), pre-swizzled bf16 B fragments,
// 8-step bf16 WMMA chain.
// ---------------------------------------------------------------------------
__global__ __launch_bounds__(256)
void fc_head(const float* __restrict__ Hfin,     // [N, 128] fp32
             const int* __restrict__ nest,
             const int* __restrict__ food,
             const __bf16* __restrict__ PFC,     // [16nt][8ks][32][16] bf16
             const float* __restrict__ bfc,      // [256]
             float* __restrict__ out) {          // [B, 256]
    const int lane = threadIdx.x & 31;
    const int wave = threadIdx.x >> 5;
    const int m0   = blockIdx.x * 16;                 // query tile
    const int nt   = blockIdx.y * 8 + wave;           // hidden-column tile
    const int lh   = lane & 15;
    const int kb   = (lane >> 4) * 8;
    const int q    = m0 + lh;
    const int ncol = nt * 16 + lh;
    const float* rowA = Hfin + (size_t)nest[q] * 128;
    const float* rowB = Hfin + (size_t)food[q] * 128;

    const __bf16* pb = PFC + (size_t)nt * (8 * 512) + (size_t)lane * 16;

    v8f acc = {};
    for (int ks = 0; ks < 8; ++ks) {
        const int kA = ks * 32 + kb;   // first 8-run
        const int kB = kA + 16;        // second 8-run (runs never cross 128)
        const float* p1 = (kA < 128) ? (rowA + kA) : (rowB + (kA - 128));
        const float* p2 = (kB < 128) ? (rowA + kB) : (rowB + (kB - 128));
        v16bf a;
        #pragma unroll
        for (int e = 0; e < 8; ++e) {
            a[e]     = (__bf16)p1[e];
            a[e + 8] = (__bf16)p2[e];
        }
        const v16bf b = *(const v16bf*)(pb + (size_t)ks * 512);
        acc = __builtin_amdgcn_wmma_f32_16x16x32_bf16(false, a, false, b,
                                                      (short)0, acc, false, false);
    }
    const float bv = bfc[ncol];
    #pragma unroll
    for (int r = 0; r < 8; ++r) {
        int m = m0 + r + kb;
        out[(size_t)m * HIDD + ncol] = fmaxf(acc[r] + bv, 0.0f);
    }
}

// ---------------------------------------------------------------------------
// Launch
// ---------------------------------------------------------------------------
extern "C" void kernel_launch(void* const* d_in, const int* in_sizes, int n_in,
                              void* d_out, int out_size, void* d_ws, size_t ws_size,
                              hipStream_t stream) {
    const float* x         = (const float*)d_in[0];
    const int*   edge_idx  = (const int*)  d_in[1];
    const int*   edge_type = (const int*)  d_in[2];
    const int*   nest      = (const int*)  d_in[3];
    const int*   food      = (const int*)  d_in[4];
    const float* W_rel1    = (const float*)d_in[5];
    const float* W_root1   = (const float*)d_in[6];
    const float* b1        = (const float*)d_in[7];
    const float* W_rel2    = (const float*)d_in[8];
    const float* W_root2   = (const float*)d_in[9];
    const float* b2        = (const float*)d_in[10];
    const float* W_fc      = (const float*)d_in[11];
    const float* b_fc      = (const float*)d_in[12];
    float* out = (float*)d_out;

    // Workspace layout (256B-aligned offsets):
    //   H    : bf16 [N, R, 128]   102,400,000 B   (reused by both layers)
    //   OUT1 : f32  [N, 128]       51,200,000 B
    //   OUT2 : f32  [N, 128]       51,200,000 B
    //   deg  : f32  [N*R]           1,600,000 B
    //   XB   : bf16 [N, 128]       25,600,000 B   (layer-1 in, then layer-2 in)
    //   PB1  : bf16 [5*128*128]       163,840 B   (fragment-swizzled weights)
    //   PB2  : bf16 [5*128*128]       163,840 B
    //   PFC  : bf16 [256*256]         131,072 B
    char* ws = (char*)d_ws;
    __bf16* H    = (__bf16*)(ws);
    float*  OUT1 = (float*)(ws + 102400000L);
    float*  OUT2 = (float*)(ws + 153600000L);
    float*  deg  = (float*)(ws + 204800000L);
    __bf16* XB   = (__bf16*)(ws + 206400000L);
    __bf16* PB1  = (__bf16*)(ws + 232000000L);
    __bf16* PB2  = (__bf16*)(ws + 232163840L);
    __bf16* PFC  = (__bf16*)(ws + 232327680L);

    const long NF  = (long)NN * FF;          // 12.8M
    const long NR  = (long)NN * RR;
    const long WN  = (long)FF * EMBD;        // 16384 per matrix

    // ---- activations to bf16 ----
    cvt_bf16<<<(int)((NF + 255) / 256), 256, 0, stream>>>(x, XB, NF);

    // ---- weights: convert + swizzle into WMMA B-fragment order ----
    for (int r = 0; r < RR; ++r) {
        pack_wmma_b<<<(int)((WN + 255) / 256), 256, 0, stream>>>(
            W_rel1 + (size_t)r * WN, PB1 + (size_t)r * WN, 4, 128, WN);
        pack_wmma_b<<<(int)((WN + 255) / 256), 256, 0, stream>>>(
            W_rel2 + (size_t)r * WN, PB2 + (size_t)r * WN, 4, 128, WN);
    }
    pack_wmma_b<<<(int)((WN + 255) / 256), 256, 0, stream>>>(
        W_root1, PB1 + (size_t)RR * WN, 4, 128, WN);
    pack_wmma_b<<<(int)((WN + 255) / 256), 256, 0, stream>>>(
        W_root2, PB2 + (size_t)RR * WN, 4, 128, WN);
    const long WFN = (long)2 * EMBD * HIDD;  // 65536
    pack_wmma_b<<<(int)((WFN + 255) / 256), 256, 0, stream>>>(
        W_fc, PFC, 8, 256, WFN);

    // ---- degrees (identical every call; recomputed for determinism) ----
    zero_f32<<<(int)((NR + 255) / 256), 256, 0, stream>>>(deg, NR);
    count_deg<<<(EE + 255) / 256, 256, 0, stream>>>(edge_idx, edge_type, deg);

    // ---- Layer 1 ----
    rgcn_transform<<<dim3(NN / 16, RR + 1), 256, 0, stream>>>(
        XB, PB1, b1, H, OUT1);
    edge_messages<<<EE / 8, 256, 0, stream>>>(edge_idx, edge_type, H, deg, OUT1);
    relu_cvt_bf16<<<(int)((NF + 255) / 256), 256, 0, stream>>>(OUT1, XB, NF);

    // ---- Layer 2 ----
    rgcn_transform<<<dim3(NN / 16, RR + 1), 256, 0, stream>>>(
        XB, PB2, b2, H, OUT2);
    edge_messages<<<EE / 8, 256, 0, stream>>>(edge_idx, edge_type, H, deg, OUT2);

    // ---- FC head ----
    fc_head<<<dim3(BB / 16, HIDD / 128), 256, 0, stream>>>(
        OUT2, nest, food, PFC, b_fc, out);
}
